// PointNet2_seg_61847529062862
// MI455X (gfx1250) — matrix operand order repaired and checked by reading, not compiled, not run on hardware
//
#include <hip/hip_runtime.h>
#include <hip/hip_bf16.h>

// ---------------------------------------------------------------------------
// PointNet++ segmentation (B=4, N=4096, K=32, CIN=16, C=128, NL=3) for
// MI455X / gfx1250. All GEMMs run on v_wmma_f32_16x16x32_bf16 with f32
// accumulation; the per-neighbor activation tensor lives only in LDS.
// ---------------------------------------------------------------------------

typedef __bf16 bf16;
typedef __bf16 v16bf __attribute__((ext_vector_type(16)));
typedef __bf16 v8bf  __attribute__((ext_vector_type(8)));
typedef float  v8f   __attribute__((ext_vector_type(8)));

#define NB  4      // batches (clouds)
#define NP  4096   // points per cloud
#define NK  32     // neighbor cap
#define NC  128    // channels

// ---------------------------------------------------------------------------
// WMMA fragment helpers (layouts per CDNA5 ISA 7.12.2, wave32)
// ---------------------------------------------------------------------------
__device__ __forceinline__ v16bf frag_cat(v8bf lo, v8bf hi) {
  return __builtin_shufflevector(lo, hi, 0,1,2,3,4,5,6,7,8,9,10,11,12,13,14,15);
}

// A (16x32, MxK), row-major LDS, 16-bit: lane&15 = row, lane>>4 selects the
// K-halves {0..7,16..23} vs {8..15,24..31}. Two contiguous 16B LDS loads.
__device__ __forceinline__ v16bf load_a_frag(const bf16* base, int row0,
                                             int ldk, int k0, int lane) {
  int m  = row0 + (lane & 15);
  int kh = (lane >> 4) << 3;                    // 0 or 8
  const bf16* p = base + m * ldk + k0 + kh;
  v8bf lo = *(const v8bf*)p;                    // K = k0+kh .. +7
  v8bf hi = *(const v8bf*)(p + 16);             // K = k0+16+kh .. +7
  return frag_cat(lo, hi);
}

// B (32x16, KxN) from a TRANSPOSED weight tile Wt[n][k] (row-major in n):
// lane&15 = column n, lane>>4 selects K-halves {0..7,16..23} vs {8..15,24..31}.
__device__ __forceinline__ v16bf load_b_frag(const bf16* baseT, int n0,
                                             int ldk, int k0, int lane) {
  int n  = n0 + (lane & 15);
  int kh = (lane >> 4) << 3;                    // 0 or 8
  const bf16* p = baseT + n * ldk + k0 + kh;
  v8bf lo = *(const v8bf*)p;
  v8bf hi = *(const v8bf*)(p + 16);
  return frag_cat(lo, hi);
}

__device__ __forceinline__ v8f wmma_bf16(v16bf a, v16bf b, v8f c) {
  // (neg_a, A, neg_b, B, c_mod, C, reuse_a, reuse_b)
  return __builtin_amdgcn_wmma_f32_16x16x32_bf16(false, a, false, b,
                                                 (short)0, c, false, false);
}

// ---------------------------------------------------------------------------
// Kernel 1: radius-limited 32-NN (self included). One thread per query point,
// candidate positions tiled through LDS. Invalid slots get idx = -1.
// ---------------------------------------------------------------------------
__global__ __launch_bounds__(256)
void knn_kernel(const float* __restrict__ pos, int* __restrict__ nidx) {
  __shared__ float spx[256], spy[256], spz[256];
  const int b   = blockIdx.x >> 4;                      // 16 blocks / cloud
  const int qil = ((blockIdx.x & 15) << 8) | threadIdx.x;
  const int qi  = b * NP + qil;
  const float qx = pos[qi * 3 + 0];
  const float qy = pos[qi * 3 + 1];
  const float qz = pos[qi * 3 + 2];

  float bd[NK]; int bj[NK];
#pragma unroll
  for (int s = 0; s < NK; ++s) { bd[s] = 3.0e38f; bj[s] = -1; }
  const float R2 = 0.01f;                               // R = 0.1

  for (int t = 0; t < NP / 256; ++t) {
    __syncthreads();
    const int cj = b * NP + t * 256 + threadIdx.x;
    spx[threadIdx.x] = pos[cj * 3 + 0];
    spy[threadIdx.x] = pos[cj * 3 + 1];
    spz[threadIdx.x] = pos[cj * 3 + 2];
    __syncthreads();
    for (int c = 0; c < 256; ++c) {
      const float dx = qx - spx[c];
      const float dy = qy - spy[c];
      const float dz = qz - spz[c];
      const float d2 = dx * dx + dy * dy + dz * dz;
      if (d2 <= R2 && d2 < bd[NK - 1]) {
        bd[NK - 1] = d2; bj[NK - 1] = b * NP + t * 256 + c;
#pragma unroll
        for (int s = NK - 1; s > 0; --s) {
          if (bd[s] < bd[s - 1]) {
            float td = bd[s]; bd[s] = bd[s - 1]; bd[s - 1] = td;
            int   tj = bj[s]; bj[s] = bj[s - 1]; bj[s - 1] = tj;
          }
        }
      }
    }
  }
#pragma unroll
  for (int s = 0; s < NK; ++s) nidx[qi * NK + s] = bj[s];
}

// ---------------------------------------------------------------------------
// Kernel 2: pad layer-0 features [16384,16] f32 -> [16384,32] bf16 (zero pad)
// ---------------------------------------------------------------------------
__global__ __launch_bounds__(256)
void prep_x0_kernel(const float* __restrict__ feat, bf16* __restrict__ x0p) {
  const int li = blockIdx.x * 256 + threadIdx.x;        // 16384*32 total
  const int r = li >> 5, c = li & 31;
  x0p[li] = (c < 16) ? (bf16)feat[r * 16 + c] : (bf16)0.0f;
}

// ---------------------------------------------------------------------------
// Kernel 3: fused PointNetConv layer.
// One workgroup = 4 points x 32 neighbors = 128 rows, full 128 output chans.
// LDS: W1t[128][KA], W2t[128][128] (transposed bf16 weights with BN-fused
// scale/shift), gathered A[128][KA], hidden H[128][128] bf16, OUT f32, then
// max-over-neighbors reduce -> Xi bf16.
// ---------------------------------------------------------------------------
template <int CIN>
__global__ __launch_bounds__(256)
void conv_kernel(const bf16* __restrict__ Xin,
                 const float* __restrict__ pos,
                 const int*  __restrict__ nidx,
                 const float* __restrict__ wa, const float* __restrict__ ba,
                 const float* __restrict__ ga, const float* __restrict__ bea,
                 const float* __restrict__ wb, const float* __restrict__ bb,
                 const float* __restrict__ gb, const float* __restrict__ beb,
                 bf16* __restrict__ Xi) {
  constexpr int KA     = (CIN + 3 + 31) & ~31;          // 32 or 160
  constexpr int ROWS_A = CIN + 3;
  constexpr int XS     = (CIN == 16) ? 32 : CIN;        // padded input stride
  constexpr int OFF_W2  = 128 * KA * 2;
  constexpr int OFF_A   = OFF_W2 + 128 * 128 * 2;
  constexpr int OFF_H   = OFF_A + 128 * KA * 2;
  constexpr int OFF_SV  = OFF_H + 128 * 128 * 2;
  constexpr int OFF_OUT = OFF_SV + 4 * 128 * 4;

  extern __shared__ char smem[];
  bf16*  W1t = (bf16*)(smem);
  bf16*  W2t = (bf16*)(smem + OFF_W2);
  bf16*  A   = (bf16*)(smem + OFF_A);
  bf16*  H   = (bf16*)(smem + OFF_H);
  float* SV  = (float*)(smem + OFF_SV);                 // s1,t1,s2,t2
  float* OUT = (float*)(smem + OFF_OUT);                // [128][128] f32

  const int tid = threadIdx.x;
  const int base_pt = blockIdx.x * 4;                   // 4 points / WG

  __builtin_prefetch(wa, 0, 3);
  __builtin_prefetch(wb, 0, 3);

  // Stage W1 transposed + padded (rows >= ROWS_A are zero).
  for (int li = tid; li < 128 * KA; li += 256) {
    const int n = li / KA, k = li % KA;
    const float v = (k < ROWS_A) ? wa[k * 128 + n] : 0.0f;
    W1t[n * KA + k] = (bf16)v;
  }
  // Stage W2 transposed.
  for (int li = tid; li < 128 * 128; li += 256) {
    const int n = li >> 7, k = li & 127;
    W2t[n * 128 + k] = (bf16)wb[k * 128 + n];
  }
  // Fused BatchNorm scale/shift: z = y*s + (b*s + beta), s = g/sqrt(1+eps).
  if (tid < 128) {
    const float s = ga[tid] * rsqrtf(1.0f + 1e-5f);
    SV[tid]       = s;
    SV[128 + tid] = bea[tid] + ba[tid] * s;
  } else {
    const int u = tid - 128;
    const float s = gb[u] * rsqrtf(1.0f + 1e-5f);
    SV[256 + u] = s;
    SV[384 + u] = beb[u] + bb[u] * s;
  }
  // Gather A: row r = (point i_local, neighbor k); cols = [x_j | pos_j-pos_i | 0].
  for (int li = tid; li < 128 * KA; li += 256) {
    const int r = li / KA, c = li % KA;
    const int i = base_pt + (r >> 5);
    const int j = nidx[i * NK + (r & 31)];
    bf16 v = (bf16)0.0f;
    if (j >= 0) {
      if (c < CIN)            v = Xin[j * XS + c];
      else if (c < CIN + 3)   v = (bf16)(pos[j * 3 + (c - CIN)] -
                                         pos[i * 3 + (c - CIN)]);
    }
    A[r * KA + c] = v;
  }
  __syncthreads();

  const int w = tid >> 5, lane = tid & 31;
  const int nlo = lane & 15, mhi = (lane >> 4) << 3;

  // GEMM1: H = LReLU(BN(A @ W1)), rows w*16..w*16+15, all 128 cols.
  {
    v8f acc[8] = {};
#pragma unroll
    for (int kt = 0; kt < KA / 32; ++kt) {
      const v16bf aF = load_a_frag(A, w * 16, KA, kt * 32, lane);
#pragma unroll
      for (int nt = 0; nt < 8; ++nt) {
        const v16bf bF = load_b_frag(W1t, nt * 16, KA, kt * 32, lane);
        acc[nt] = wmma_bf16(aF, bF, acc[nt]);
      }
    }
#pragma unroll
    for (int nt = 0; nt < 8; ++nt) {
      const int n = nt * 16 + nlo;
      const float s = SV[n], t = SV[128 + n];
#pragma unroll
      for (int r = 0; r < 8; ++r) {
        const int m = w * 16 + mhi + r;
        float z = acc[nt][r] * s + t;
        z = (z >= 0.0f) ? z : 0.2f * z;
        H[m * 128 + n] = (bf16)z;
      }
    }
  }
  __syncthreads();

  // GEMM2: OUT = LReLU(BN(H @ W2)) in f32.
  {
    v8f acc[8] = {};
#pragma unroll
    for (int kt = 0; kt < 4; ++kt) {
      const v16bf aF = load_a_frag(H, w * 16, 128, kt * 32, lane);
#pragma unroll
      for (int nt = 0; nt < 8; ++nt) {
        const v16bf bF = load_b_frag(W2t, nt * 16, 128, kt * 32, lane);
        acc[nt] = wmma_bf16(aF, bF, acc[nt]);
      }
    }
#pragma unroll
    for (int nt = 0; nt < 8; ++nt) {
      const int n = nt * 16 + nlo;
      const float s = SV[256 + n], t = SV[384 + n];
#pragma unroll
      for (int r = 0; r < 8; ++r) {
        const int m = w * 16 + mhi + r;
        float z = acc[nt][r] * s + t;
        z = (z >= 0.0f) ? z : 0.2f * z;
        OUT[m * 128 + n] = z;
      }
    }
  }
  __syncthreads();

  // Max over valid neighbors -> Xi[point][chan].
  for (int o = tid; o < 4 * 128; o += 256) {
    const int iL = o >> 7, n = o & 127;
    const int i = base_pt + iL;
    float mx = -3.0e38f;
#pragma unroll 4
    for (int k = 0; k < NK; ++k) {
      if (nidx[i * NK + k] >= 0)
        mx = fmaxf(mx, OUT[(iL * NK + k) * 128 + n]);
    }
    Xi[i * 128 + n] = (bf16)mx;
  }
}

// ---------------------------------------------------------------------------
// Kernel 4: fused per-point linears + residual transform:
//   out = relu(Xi@lw1 + lb1)@lw2 + lb2 + X@lt + ltb
// The residual GEMM accumulates into the same WMMA accumulators.
// CX = padded K-dim of lt (32 or 128); RK = real rows of lt (16 or 128).
// ---------------------------------------------------------------------------
template <int CX, int RK>
__global__ __launch_bounds__(256)
void linear_kernel(const bf16* __restrict__ Xi, const bf16* __restrict__ X,
                   const float* __restrict__ lw1, const float* __restrict__ lb1,
                   const float* __restrict__ lw2, const float* __restrict__ lb2,
                   const float* __restrict__ ltw, const float* __restrict__ ltb,
                   bf16* __restrict__ Xn, float* __restrict__ OutF) {
  extern __shared__ char smem[];
  bf16* W1t = (bf16*)(smem);                            // [128][128]
  bf16* W2t = (bf16*)(smem + 32768);                    // [128][128]
  bf16* Ltt = (bf16*)(smem + 65536);                    // [128][CX]
  bf16* A   = (bf16*)(smem + 65536 + 256 * CX);         // [128][128]
  bf16* Xt  = (bf16*)(smem + 98304 + 256 * CX);         // [128][CX]
  bf16* H   = (bf16*)(smem + 98304 + 512 * CX);         // [128][128]

  const int tid  = threadIdx.x;
  const int base = blockIdx.x * 128;                    // 128 points / WG

  __builtin_prefetch(lw1, 0, 3);
  __builtin_prefetch(lw2, 0, 3);
  __builtin_prefetch(ltw, 0, 3);

  for (int li = tid; li < 128 * 128; li += 256) {
    const int n = li >> 7, k = li & 127;
    W1t[li] = (bf16)lw1[k * 128 + n];
    W2t[li] = (bf16)lw2[k * 128 + n];
  }
  for (int li = tid; li < 128 * CX; li += 256) {
    const int n = li / CX, k = li % CX;
    Ltt[li] = (k < RK) ? (bf16)ltw[k * 128 + n] : (bf16)0.0f;
  }
  {
    const uint4* s1 = (const uint4*)(Xi + (size_t)base * 128);
    uint4* d1 = (uint4*)A;
    for (int li = tid; li < 128 * 128 / 8; li += 256) d1[li] = s1[li];
    const uint4* s2 = (const uint4*)(X + (size_t)base * CX);
    uint4* d2 = (uint4*)Xt;
    for (int li = tid; li < 128 * CX / 8; li += 256) d2[li] = s2[li];
  }
  __syncthreads();

  const int w = tid >> 5, lane = tid & 31;
  const int nlo = lane & 15, mhi = (lane >> 4) << 3;

  // H = relu(A @ W1 + lb1)
  {
    v8f acc[8] = {};
#pragma unroll
    for (int kt = 0; kt < 4; ++kt) {
      const v16bf aF = load_a_frag(A, w * 16, 128, kt * 32, lane);
#pragma unroll
      for (int nt = 0; nt < 8; ++nt) {
        const v16bf bF = load_b_frag(W1t, nt * 16, 128, kt * 32, lane);
        acc[nt] = wmma_bf16(aF, bF, acc[nt]);
      }
    }
#pragma unroll
    for (int nt = 0; nt < 8; ++nt) {
      const int n = nt * 16 + nlo;
      const float bv = lb1[n];
#pragma unroll
      for (int r = 0; r < 8; ++r) {
        const int m = w * 16 + mhi + r;
        const float z = fmaxf(acc[nt][r] + bv, 0.0f);
        H[m * 128 + n] = (bf16)z;
      }
    }
  }
  __syncthreads();

  // acc = H @ W2 + Xt @ Lt ; then + lb2 + ltb
  {
    v8f acc[8] = {};
#pragma unroll
    for (int kt = 0; kt < 4; ++kt) {
      const v16bf aF = load_a_frag(H, w * 16, 128, kt * 32, lane);
#pragma unroll
      for (int nt = 0; nt < 8; ++nt) {
        const v16bf bF = load_b_frag(W2t, nt * 16, 128, kt * 32, lane);
        acc[nt] = wmma_bf16(aF, bF, acc[nt]);
      }
    }
#pragma unroll
    for (int kt = 0; kt < CX / 32; ++kt) {
      const v16bf aF = load_a_frag(Xt, w * 16, CX, kt * 32, lane);
#pragma unroll
      for (int nt = 0; nt < 8; ++nt) {
        const v16bf bF = load_b_frag(Ltt, nt * 16, CX, kt * 32, lane);
        acc[nt] = wmma_bf16(aF, bF, acc[nt]);
      }
    }
#pragma unroll
    for (int nt = 0; nt < 8; ++nt) {
      const int n = nt * 16 + nlo;
      const float bv = lb2[n] + ltb[n];
#pragma unroll
      for (int r = 0; r < 8; ++r) {
        const int m = w * 16 + mhi + r;
        const float z = acc[nt][r] + bv;
        const int gi = (base + m) * 128 + n;
        Xn[gi] = (bf16)z;
        if (OutF) OutF[gi] = z;
      }
    }
  }
}

// ---------------------------------------------------------------------------
// Host launcher
// ---------------------------------------------------------------------------
extern "C" void kernel_launch(void* const* d_in, const int* in_sizes, int n_in,
                              void* d_out, int out_size, void* d_ws, size_t ws_size,
                              hipStream_t stream) {
  const float* positions = (const float*)d_in[0];
  const float* features  = (const float*)d_in[1];
  const float* w0a  = (const float*)d_in[3];
  const float* b0a  = (const float*)d_in[4];
  const float* g0a  = (const float*)d_in[5];
  const float* be0a = (const float*)d_in[6];
  const float* w0b  = (const float*)d_in[7];
  const float* b0b  = (const float*)d_in[8];
  const float* g0b  = (const float*)d_in[9];
  const float* be0b = (const float*)d_in[10];
  const float* wa   = (const float*)d_in[11];
  const float* ba   = (const float*)d_in[12];
  const float* ga   = (const float*)d_in[13];
  const float* bea  = (const float*)d_in[14];
  const float* wb   = (const float*)d_in[15];
  const float* bbp  = (const float*)d_in[16];
  const float* gb   = (const float*)d_in[17];
  const float* beb  = (const float*)d_in[18];
  const float* lw1  = (const float*)d_in[19];
  const float* lb1  = (const float*)d_in[20];
  const float* lw2  = (const float*)d_in[21];
  const float* lb2  = (const float*)d_in[22];
  const float* lt0w = (const float*)d_in[23];
  const float* lt0b = (const float*)d_in[24];
  const float* ltw  = (const float*)d_in[25];
  const float* ltb  = (const float*)d_in[26];

  char* ws = (char*)d_ws;
  int*  nidx  = (int*)ws;                               // 2 MB
  bf16* x0p   = (bf16*)(ws + (size_t)(2u << 20));       // 1 MB  [16384][32]
  bf16* xcur  = (bf16*)(ws + (size_t)(3u << 20));       // 4 MB  [16384][128]
  bf16* xi    = (bf16*)(ws + (size_t)(7u << 20));       // 4 MB
  bf16* xnext = (bf16*)(ws + (size_t)(11u << 20));      // 4 MB

  prep_x0_kernel<<<2048, 256, 0, stream>>>(features, x0p);
  knn_kernel<<<64, 256, 0, stream>>>(positions, nidx);

  // --- layer 0 (CIN=16, KA=32) ---
  conv_kernel<16><<<4096, 256, 149504, stream>>>(
      x0p, positions, nidx, w0a, b0a, g0a, be0a, w0b, b0b, g0b, be0b, xi);
  linear_kernel<32, 16><<<128, 256, 147456, stream>>>(
      xi, x0p, lw1, lb1, lw2, lb2, lt0w, lt0b, xcur, nullptr);

  // --- layer 1 (CIN=128, KA=160) ---
  conv_kernel<128><<<4096, 256, 215040, stream>>>(
      xcur, positions, nidx,
      wa, ba, ga, bea, wb, bbp, gb, beb, xi);
  linear_kernel<128, 128><<<128, 256, 196608, stream>>>(
      xi, xcur, lw1 + 16384, lb1 + 128, lw2 + 16384, lb2 + 128,
      ltw, ltb, xnext, nullptr);

  // --- layer 2 (CIN=128, KA=160), final f32 output ---
  conv_kernel<128><<<4096, 256, 215040, stream>>>(
      xnext, positions, nidx,
      wa + 131 * 128, ba + 128, ga + 128, bea + 128,
      wb + 16384, bbp + 128, gb + 128, beb + 128, xi);
  linear_kernel<128, 128><<<128, 256, 196608, stream>>>(
      xi, xnext, lw1 + 2 * 16384, lb1 + 256, lw2 + 2 * 16384, lb2 + 256,
      ltw + 16384, ltb + 128, xcur, (float*)d_out);
}
